// StackedBiGRU_56109452755060
// MI455X (gfx1250) — matrix operand order repaired and checked by reading, not compiled, not run on hardware
//
#include <hip/hip_runtime.h>

// ---------------------------------------------------------------------------
// StackedBiGRU for MI455X (gfx1250, wave32, WMMA, async global->LDS)
//   T=8192, DIN=64, H=512, HS=512, S=64 segments, C=48
// Strategy:
//   * All input projections (x@Wih^T + b) done as bf16 WMMA GEMMs (f32 accum),
//     software-pipelined with GLOBAL_LOAD_ASYNC_TO_LDS_B128 double buffering.
//   * Recurrence: 2 dirs x 8 WGs; each WG holds a 192x512 bf16 slice of Whh in
//     LDS (192KB of the 320KB WGP LDS), computes 64 hidden outputs/step,
//     exchanges h through L2 with a device-scope atomic barrier (triple-
//     buffered h so one barrier per step suffices).
//   * Pooling / MLP are tiny VALU kernels.
// ---------------------------------------------------------------------------

#define T_SEQ 8192
#define H_DIM 512
#define G3    1536           // 3*H
#define NWG_PER_DIR 8
#define NWG_TOTAL   16       // 2 dirs * 8

typedef __attribute__((ext_vector_type(16))) __bf16 v16bf;
typedef __attribute__((ext_vector_type(8)))  float  v8f;

union Frag16 { v16bf v; unsigned int u[8]; };

__device__ __forceinline__ unsigned short f2bf(float x) {
  unsigned int u = __float_as_uint(x);
  unsigned int r = u + 0x7FFFu + ((u >> 16) & 1u);   // round-to-nearest-even
  return (unsigned short)(r >> 16);
}
__device__ __forceinline__ float bf_lo(unsigned int w) { return __uint_as_float(w << 16); }
__device__ __forceinline__ float bf_hi(unsigned int w) { return __uint_as_float(w & 0xFFFF0000u); }
__device__ __forceinline__ float sigmoidf_(float x) { return 1.0f / (1.0f + __expf(-x)); }

// LDS byte offset of a __shared__ object: the LDS aperture occupies flat
// address bits [63:32], so the low 32 bits are the intra-allocation offset.
__device__ __forceinline__ unsigned lds_offset(const void* p) {
  return (unsigned)(unsigned long long)(uintptr_t)p;
}
// CDNA5 async DMA: memory -> LDS, 16 bytes per lane, tracked by ASYNCcnt.
__device__ __forceinline__ void async_load_b128(unsigned lds_off,
                                                const void* gaddr) {
  asm volatile("global_load_async_to_lds_b128 %0, %1, off"
               :: "v"(lds_off), "v"((unsigned long long)(uintptr_t)gaddr)
               : "memory");
}
__device__ __forceinline__ void wait_async0() {
  asm volatile("s_wait_asynccnt 0" ::: "memory");
}

// ---------------------------------------------------------------------------
// f32 -> bf16 conversion (grid-stride)
// ---------------------------------------------------------------------------
__global__ void f32_to_bf16_kernel(const float* __restrict__ src,
                                   unsigned short* __restrict__ dst, long n) {
  long i = (long)blockIdx.x * blockDim.x + threadIdx.x;
  long stride = (long)gridDim.x * blockDim.x;
  for (; i < n; i += stride) dst[i] = f2bf(src[i]);
}

__global__ void zero_bar_kernel(unsigned int* bar) {
  if (threadIdx.x < 8) bar[threadIdx.x] = 0u;
}

// ---------------------------------------------------------------------------
// WMMA GEMM:  C[z][M x N] = A[M x K](bf16) @ B[z][N x K]^T(bf16) + bias[z][N]
// Tile 128x128x32, 256 threads = 8 waves (2x4), each wave: 4x2 WMMA tiles.
// Double-buffered LDS tiles filled by async global->LDS DMA; next K-tile's
// loads are issued before the current tile's WMMAs so the matrix pipe hides
// all memory latency.
// Fragment layouts follow CDNA5 ISA 7.12.2:
//   A 16x32 bf16: lane l (l<16: M=l, K half 0; l>=16: M=l-16, K half +8);
//                 pairs p<4 -> K=2p+half*8, p>=4 -> K=16+2(p-4)+half*8.
//   B 32x16 bf16: lane n = column; K per pair p and lane half exactly as for A.
// ---------------------------------------------------------------------------
__global__ void __launch_bounds__(256)
wmma_gemm_bias(const unsigned short* __restrict__ A,
               const unsigned short* __restrict__ B,
               const float* __restrict__ bias,
               float* __restrict__ C,
               int M, int N, int K,
               long long strideB, long long strideC, long long strideBias) {
  __shared__ __align__(16) unsigned short As[2][128 * 32];
  __shared__ __align__(16) unsigned short Bs[2][128 * 32];

  const int z = blockIdx.z;
  const unsigned short* Bz = B + (size_t)z * strideB;
  float* Cz = C + (size_t)z * strideC;
  const float* biasz = bias + (size_t)z * strideBias;

  const int tid  = threadIdx.x;
  const int wave = tid >> 5;
  const int lane = tid & 31;
  const int wrow = wave >> 2;       // 0..1 -> 64-row band
  const int wcol = wave & 3;        // 0..3 -> 32-col band
  const int half = lane >> 4;
  const int lr   = lane & 15;
  const int rowBase = blockIdx.y * 128;
  const int colBase = blockIdx.x * 128;

  const uint4 zero4 = {0u, 0u, 0u, 0u};

  // issue async DMA for one 128x32 A-tile and B-tile into buffer `buf`
  auto issue_tile = [&](int k0, int buf) {
#pragma unroll
    for (int it = 0; it < 2; ++it) {
      int li = tid + it * 256;            // 0..511
      int r  = li >> 2;                   // 0..127
      int c  = (li & 3) << 3;             // 0,8,16,24 (bf16 units)
      int gr = rowBase + r;
      unsigned short* adst = &As[buf][r * 32 + c];
      if (gr < M)
        async_load_b128(lds_offset(adst), A + (size_t)gr * K + k0 + c);
      else
        *(uint4*)adst = zero4;
      int gc = colBase + r;               // B row == output column
      unsigned short* bdst = &Bs[buf][r * 32 + c];
      if (gc < N)
        async_load_b128(lds_offset(bdst), Bz + (size_t)gc * K + k0 + c);
      else
        *(uint4*)bdst = zero4;
    }
  };

  v8f acc[4][2];
  const v8f zf = {0.f, 0.f, 0.f, 0.f, 0.f, 0.f, 0.f, 0.f};
#pragma unroll
  for (int i = 0; i < 4; ++i)
#pragma unroll
    for (int j = 0; j < 2; ++j) acc[i][j] = zf;

  int buf = 0;
  issue_tile(0, 0);
  wait_async0();
  __syncthreads();

  for (int k0 = 0; k0 < K; k0 += 32) {
    // start DMA of the next K-tile into the other buffer (overlaps WMMAs)
    if (k0 + 32 < K) issue_tile(k0 + 32, buf ^ 1);

    Frag16 afr[4], bfr[2];
#pragma unroll
    for (int jt = 0; jt < 2; ++jt) {
      int col = wcol * 32 + jt * 16 + lr;
#pragma unroll
      for (int p = 0; p < 8; ++p) {
        int kk = (p < 4 ? 2 * p : 16 + 2 * (p - 4)) + half * 8;  // B layout
        bfr[jt].u[p] = *(const unsigned int*)(&Bs[buf][col * 32 + kk]);
      }
    }
#pragma unroll
    for (int it = 0; it < 4; ++it) {
      int row = wrow * 64 + it * 16 + lr;
#pragma unroll
      for (int p = 0; p < 8; ++p) {
        int kk = (p < 4 ? 2 * p : 16 + 2 * (p - 4)) + half * 8;  // A layout
        afr[it].u[p] = *(const unsigned int*)(&As[buf][row * 32 + kk]);
      }
    }

#pragma unroll
    for (int it = 0; it < 4; ++it)
#pragma unroll
      for (int jt = 0; jt < 2; ++jt)
        acc[it][jt] = __builtin_amdgcn_wmma_f32_16x16x32_bf16(
            false, afr[it].v, false, bfr[jt].v, (short)0, acc[it][jt],
            false, false);

    wait_async0();        // next tile's DMA complete (ASYNCcnt == 0)
    __syncthreads();      // everyone done reading current buffer
    buf ^= 1;
  }

  // epilogue: C/D 16x16 layout -> row = v + half*8, col = lr
#pragma unroll
  for (int it = 0; it < 4; ++it) {
#pragma unroll
    for (int jt = 0; jt < 2; ++jt) {
      int col = colBase + wcol * 32 + jt * 16 + lr;
      float bv = biasz[col];
#pragma unroll
      for (int v = 0; v < 8; ++v) {
        int row = rowBase + wrow * 64 + it * 16 + half * 8 + v;
        if (row < M) Cz[(size_t)row * N + col] = acc[it][jt][v] + bv;
      }
    }
  }
}

// ---------------------------------------------------------------------------
// Inter-workgroup barrier (16 WGs, monotonic counter, device scope)
// ---------------------------------------------------------------------------
__device__ __forceinline__ void grid_barrier(unsigned int* bar, unsigned int target) {
  __syncthreads();
  if (threadIdx.x == 0) {
    __hip_atomic_fetch_add(bar, 1u, __ATOMIC_RELEASE, __HIP_MEMORY_SCOPE_AGENT);
    while (__hip_atomic_load(bar, __ATOMIC_ACQUIRE, __HIP_MEMORY_SCOPE_AGENT) < target)
      __builtin_amdgcn_s_sleep(1);
  }
  __syncthreads();
}

// ---------------------------------------------------------------------------
// GRU scan. Grid = 16 blocks (dir = bid/8, wg = bid%8), 256 threads.
// Each WG owns hidden slice j in [wg*64, wg*64+64), holds the 3 gate rows for
// that slice of Whh (192 x 512 bf16 = 192KB) in dynamic LDS.
// gi:   [2][T][1536] f32 (includes b_ih)      whh: [2][1536][512] bf16
// bhh:  [2][1536]    h0: [2][512]             hbuf: [2][3][512] (triple buffer)
// ---------------------------------------------------------------------------
__global__ void __launch_bounds__(256)
gru_scan(const float* __restrict__ gi,
         const unsigned short* __restrict__ whh,
         const float* __restrict__ bhh,
         const float* __restrict__ h0,
         float* __restrict__ hbuf,
         unsigned int* __restrict__ bar,
         int T,
         float* __restrict__ out_f32,
         unsigned short* __restrict__ out_bf16,
         int outStride) {
  extern __shared__ unsigned char smem_raw[];
  unsigned short* w_s = (unsigned short*)smem_raw;            // 3*64*512 bf16
  float* h_s = (float*)(smem_raw + 3 * 64 * 512 * 2);         // 512 f32
  float* bb  = h_s + 512;                                     // 192 f32

  const int tid = threadIdx.x;
  const int dir = blockIdx.x >> 3;
  const int wg  = blockIdx.x & 7;
  const int j0  = wg * 64;

  // load Whh slice (rows g*512 + j0 .. j0+63, g=0..2) into LDS as bf16
  const unsigned short* wsrc = whh + (size_t)dir * G3 * H_DIM;
  for (int idx = tid; idx < 3 * 64 * 512; idx += 256) {
    int g = idx >> 15;              // /(64*512)
    int r = (idx >> 9) & 63;
    int k = idx & 511;
    w_s[idx] = wsrc[(size_t)(g * 512 + j0 + r) * 512 + k];
  }
  for (int i = tid; i < 192; i += 256) {
    int g = i >> 6, r = i & 63;
    bb[i] = bhh[dir * G3 + g * 512 + j0 + r];
  }

  float* hb = hbuf + dir * 3 * 512;
  for (int i = tid; i < 64; i += 256)
    hb[j0 + i] = h0[dir * 512 + j0 + i];        // buffer 0 = h0

  grid_barrier(bar, NWG_TOTAL);                 // init round

  const int jl = tid >> 2;         // 0..63  local hidden index
  const int ks = tid & 3;          // k-slice (128 each)
  const int j  = j0 + jl;

  int cur = 0;
  for (int s = 0; s < T; ++s) {
    const int t = (dir == 0) ? s : (T - 1 - s);

    // stage full h into LDS
    for (int i = tid; i < 512; i += 256) h_s[i] = hb[cur * 512 + i];
    __syncthreads();

    // prefetch next step's gi for our slice (global_prefetch_b8)
    if (s + 1 < T && ks == 0) {
      int tn = (dir == 0) ? (s + 1) : (T - 2 - s);
      __builtin_prefetch(gi + ((size_t)dir * T + tn) * G3 + j, 0, 1);
    }

    float ar = 0.f, az = 0.f, an = 0.f;
    const unsigned int* wr = (const unsigned int*)(w_s + (size_t)jl * 512);
    const unsigned int* wz = (const unsigned int*)(w_s + (size_t)(64 + jl) * 512);
    const unsigned int* wn = (const unsigned int*)(w_s + (size_t)(128 + jl) * 512);
    const int kb = ks * 64;        // in uint (2 bf16) units
#pragma unroll 8
    for (int ku = kb; ku < kb + 64; ++ku) {
      float hv0 = h_s[2 * ku], hv1 = h_s[2 * ku + 1];
      unsigned int r2 = wr[ku], z2 = wz[ku], n2 = wn[ku];
      ar = fmaf(bf_lo(r2), hv0, fmaf(bf_hi(r2), hv1, ar));
      az = fmaf(bf_lo(z2), hv0, fmaf(bf_hi(z2), hv1, az));
      an = fmaf(bf_lo(n2), hv0, fmaf(bf_hi(n2), hv1, an));
    }
    // reduce across the 4 k-slices (contiguous lanes in one wave32)
    ar += __shfl_xor(ar, 1); ar += __shfl_xor(ar, 2);
    az += __shfl_xor(az, 1); az += __shfl_xor(az, 2);
    an += __shfl_xor(an, 1); an += __shfl_xor(an, 2);

    if (ks == 0) {
      const float* git = gi + ((size_t)dir * T + t) * G3;
      float ghr = ar + bb[jl];
      float ghz = az + bb[64 + jl];
      float ghn = an + bb[128 + jl];
      float r = sigmoidf_(git[j] + ghr);
      float zz = sigmoidf_(git[512 + j] + ghz);
      float n = tanhf(git[1024 + j] + r * ghn);
      float hnew = (1.f - zz) * n + zz * h_s[j];
      hb[((cur + 1) % 3) * 512 + j] = hnew;
      size_t oidx = (size_t)t * outStride + dir * 512 + j;
      if (out_f32)  out_f32[oidx] = hnew;
      if (out_bf16) out_bf16[oidx] = f2bf(hnew);
    }
    grid_barrier(bar, (unsigned int)NWG_TOTAL * (s + 2));
    cur = (cur + 1) % 3;
  }
}

// ---------------------------------------------------------------------------
// Segment pooling: y1[8192][1024] -> pooled[64][2048] (max | mean)
// ---------------------------------------------------------------------------
__global__ void pool_kernel(const float* __restrict__ y,
                            const int* __restrict__ seg,
                            float* __restrict__ pooled,
                            unsigned short* __restrict__ pooled_bf) {
  const int s = blockIdx.x;
  const int start = seg[2 * s], end = seg[2 * s + 1];
  const float inv = 1.0f / (float)(end - start);
  for (int d = threadIdx.x; d < 1024; d += 256) {
    float mx = -3.402823e38f, sum = 0.f;
    for (int t = start; t < end; ++t) {
      float v = y[(size_t)t * 1024 + d];
      mx = fmaxf(mx, v);
      sum += v;
    }
    float mean = sum * inv;
    pooled[s * 2048 + d] = mx;
    pooled[s * 2048 + 1024 + d] = mean;
    pooled_bf[s * 2048 + d] = f2bf(mx);
    pooled_bf[s * 2048 + 1024 + d] = f2bf(mean);
  }
}

// ---------------------------------------------------------------------------
// MLP head: s1[64][1024] -> fc1(120,lrelu) -> fc2(80,lrelu) -> out(48)
// ---------------------------------------------------------------------------
__global__ void __launch_bounds__(128)
mlp_kernel(const float* __restrict__ S1,
           const float* __restrict__ w1, const float* __restrict__ b1,
           const float* __restrict__ w2, const float* __restrict__ b2,
           const float* __restrict__ w3, const float* __restrict__ b3,
           float* __restrict__ out) {
  __shared__ float xin[1024];
  __shared__ float h1[120];
  __shared__ float h2[80];
  const int s = blockIdx.x, tid = threadIdx.x;
  for (int i = tid; i < 1024; i += 128) xin[i] = S1[(size_t)s * 1024 + i];
  __syncthreads();
  for (int o = tid; o < 120; o += 128) {
    float a = b1[o];
    const float* w = w1 + (size_t)o * 1024;
    for (int k = 0; k < 1024; ++k) a = fmaf(w[k], xin[k], a);
    h1[o] = a > 0.f ? a : 0.01f * a;
  }
  __syncthreads();
  for (int o = tid; o < 80; o += 128) {
    float a = b2[o];
    const float* w = w2 + (size_t)o * 120;
    for (int k = 0; k < 120; ++k) a = fmaf(w[k], h1[k], a);
    h2[o] = a > 0.f ? a : 0.01f * a;
  }
  __syncthreads();
  for (int o = tid; o < 48; o += 128) {
    float a = b3[o];
    const float* w = w3 + (size_t)o * 80;
    for (int k = 0; k < 80; ++k) a = fmaf(w[k], h2[k], a);
    out[(size_t)s * 48 + o] = a;
  }
}

// ---------------------------------------------------------------------------
// Host orchestration
// ---------------------------------------------------------------------------
extern "C" void kernel_launch(void* const* d_in, const int* in_sizes, int n_in,
                              void* d_out, int out_size, void* d_ws, size_t ws_size,
                              hipStream_t stream) {
  const float* x      = (const float*)d_in[0];
  const int*   seg    = (const int*)d_in[1];
  const float* h0     = (const float*)d_in[2];
  const float* sh0    = (const float*)d_in[3];
  const float* w_ih0  = (const float*)d_in[4];
  const float* w_hh0  = (const float*)d_in[5];
  const float* b_ih0  = (const float*)d_in[6];
  const float* b_hh0  = (const float*)d_in[7];
  const float* w_ih1  = (const float*)d_in[8];
  const float* w_hh1  = (const float*)d_in[9];
  const float* b_ih1  = (const float*)d_in[10];
  const float* b_hh1  = (const float*)d_in[11];
  const float* sw_ih0 = (const float*)d_in[12];
  const float* sw_hh0 = (const float*)d_in[13];
  const float* sb_ih0 = (const float*)d_in[14];
  const float* sb_hh0 = (const float*)d_in[15];
  const float* sw_ih1 = (const float*)d_in[16];
  const float* sw_hh1 = (const float*)d_in[17];
  const float* sb_ih1 = (const float*)d_in[18];
  const float* sb_hh1 = (const float*)d_in[19];
  const float* fc1w   = (const float*)d_in[20];
  const float* fc1b   = (const float*)d_in[21];
  const float* fc2w   = (const float*)d_in[22];
  const float* fc2b   = (const float*)d_in[23];
  const float* outw   = (const float*)d_in[24];
  const float* outb   = (const float*)d_in[25];
  float* out = (float*)d_out;

  // ---- workspace carve ----
  char* p = (char*)d_ws;
  auto alloc = [&](size_t bytes) -> void* {
    void* r = (void*)p;
    p += (bytes + 255) & ~(size_t)255;
    return r;
  };
  unsigned short* x_bf     = (unsigned short*)alloc((size_t)T_SEQ * 64 * 2);
  unsigned short* wih0_bf  = (unsigned short*)alloc((size_t)2 * G3 * 64 * 2);
  unsigned short* wih1_bf  = (unsigned short*)alloc((size_t)2 * G3 * 1024 * 2);
  unsigned short* swih0_bf = (unsigned short*)alloc((size_t)2 * G3 * 2048 * 2);
  unsigned short* swih1_bf = (unsigned short*)alloc((size_t)2 * G3 * 1024 * 2);
  unsigned short* whh0_bf  = (unsigned short*)alloc((size_t)2 * G3 * 512 * 2);
  unsigned short* whh1_bf  = (unsigned short*)alloc((size_t)2 * G3 * 512 * 2);
  unsigned short* swhh0_bf = (unsigned short*)alloc((size_t)2 * G3 * 512 * 2);
  unsigned short* swhh1_bf = (unsigned short*)alloc((size_t)2 * G3 * 512 * 2);
  float*          gi       = (float*)alloc((size_t)2 * T_SEQ * G3 * 4); // reused 4x
  unsigned short* y0_bf    = (unsigned short*)alloc((size_t)T_SEQ * 1024 * 2);
  float*          y1       = (float*)alloc((size_t)T_SEQ * 1024 * 4);
  float*          pooled   = (float*)alloc((size_t)64 * 2048 * 4);
  unsigned short* pooled_bf= (unsigned short*)alloc((size_t)64 * 2048 * 2);
  unsigned short* s0_bf    = (unsigned short*)alloc((size_t)64 * 1024 * 2);
  float*          s1       = (float*)alloc((size_t)64 * 1024 * 4);
  float*          hbuf     = (float*)alloc((size_t)2 * 3 * 512 * 4);
  unsigned int*   bar      = (unsigned int*)alloc(256);

  const size_t SCAN_LDS = (size_t)3 * 64 * 512 * 2 + 512 * 4 + 192 * 4; // ~195KB

  auto conv = [&](const float* src, unsigned short* dst, long n) {
    int blocks = (int)((n + 255) / 256);
    if (blocks > 2048) blocks = 2048;
    f32_to_bf16_kernel<<<blocks, 256, 0, stream>>>(src, dst, n);
  };

  // ---- precision conversions ----
  conv(x,      x_bf,     (long)T_SEQ * 64);
  conv(w_ih0,  wih0_bf,  (long)2 * G3 * 64);
  conv(w_ih1,  wih1_bf,  (long)2 * G3 * 1024);
  conv(sw_ih0, swih0_bf, (long)2 * G3 * 2048);
  conv(sw_ih1, swih1_bf, (long)2 * G3 * 1024);
  conv(w_hh0,  whh0_bf,  (long)2 * G3 * 512);
  conv(w_hh1,  whh1_bf,  (long)2 * G3 * 512);
  conv(sw_hh0, swhh0_bf, (long)2 * G3 * 512);
  conv(sw_hh1, swhh1_bf, (long)2 * G3 * 512);
  zero_bar_kernel<<<1, 32, 0, stream>>>(bar);

  dim3 blk(256);
  // ---- layer 0: gi = x @ Wih0^T + b_ih0, both dirs ----
  wmma_gemm_bias<<<dim3(G3 / 128, T_SEQ / 128, 2), blk, 0, stream>>>(
      x_bf, wih0_bf, b_ih0, gi, T_SEQ, G3, 64,
      (long long)G3 * 64, (long long)T_SEQ * G3, (long long)G3);
  gru_scan<<<NWG_TOTAL, 256, SCAN_LDS, stream>>>(
      gi, whh0_bf, b_hh0, h0, hbuf, bar + 0, T_SEQ, nullptr, y0_bf, 1024);

  // ---- layer 1: gi = y0 @ Wih1^T + b_ih1 ----
  wmma_gemm_bias<<<dim3(G3 / 128, T_SEQ / 128, 2), blk, 0, stream>>>(
      y0_bf, wih1_bf, b_ih1, gi, T_SEQ, G3, 1024,
      (long long)G3 * 1024, (long long)T_SEQ * G3, (long long)G3);
  gru_scan<<<NWG_TOTAL, 256, SCAN_LDS, stream>>>(
      gi, whh1_bf, b_hh1, h0 + 2 * 512, hbuf, bar + 1, T_SEQ, y1, nullptr, 1024);

  // ---- segment pooling ----
  pool_kernel<<<64, 256, 0, stream>>>(y1, seg, pooled, pooled_bf);

  // ---- segment GRU layer 0 (T=64, K=2048) ----
  wmma_gemm_bias<<<dim3(G3 / 128, 1, 2), blk, 0, stream>>>(
      pooled_bf, swih0_bf, sb_ih0, gi, 64, G3, 2048,
      (long long)G3 * 2048, (long long)64 * G3, (long long)G3);
  gru_scan<<<NWG_TOTAL, 256, SCAN_LDS, stream>>>(
      gi, swhh0_bf, sb_hh0, sh0, hbuf, bar + 2, 64, nullptr, s0_bf, 1024);

  // ---- segment GRU layer 1 (T=64, K=1024) ----
  wmma_gemm_bias<<<dim3(G3 / 128, 1, 2), blk, 0, stream>>>(
      s0_bf, swih1_bf, sb_ih1, gi, 64, G3, 1024,
      (long long)G3 * 1024, (long long)64 * G3, (long long)G3);
  gru_scan<<<NWG_TOTAL, 256, SCAN_LDS, stream>>>(
      gi, swhh1_bf, sb_hh1, sh0 + 2 * 512, hbuf, bar + 3, 64, s1, nullptr, 1024);

  // ---- MLP head ----
  mlp_kernel<<<64, 128, 0, stream>>>(s1, fc1w, fc1b, fc2w, fc2b, outw, outb, out);
}
